// M_1803886264900
// MI455X (gfx1250) — matrix operand order repaired and checked by reading, not compiled, not run on hardware
//
#include <hip/hip_runtime.h>
#include <hip/hip_bf16.h>
#include <math.h>

// ---------------------------------------------------------------------------
// Shapes: M = 64*14*14 = 12544, K = 512, N = 2048
// ---------------------------------------------------------------------------
#define MROWS 12544
#define KDIM  512
#define NDIM  2048

typedef __attribute__((ext_vector_type(8)))  _Float16 v8h;
typedef __attribute__((ext_vector_type(16))) _Float16 v16h;
typedef __attribute__((ext_vector_type(8)))  float    v8f;

union AFrag { v16h v; v8h h[2]; };

// ---------------------------------------------------------------------------
// Kernel 1: residual add + LayerNorm(512) -> f16 row-major activations
// one block (512 threads, 16 waves) per row
// ---------------------------------------------------------------------------
__global__ __launch_bounds__(512) void ln_kernel(
    const float* __restrict__ x1, const float* __restrict__ x2,
    const float* __restrict__ gamma, const float* __restrict__ beta,
    _Float16* __restrict__ xn)
{
    const int row = blockIdx.x;
    const int c   = threadIdx.x;
    const size_t idx = (size_t)row * KDIM + c;

    float v = x1[idx] + x2[idx];

    // wave32 reduction
    float s = v, s2 = v * v;
    #pragma unroll
    for (int off = 16; off > 0; off >>= 1) {
        s  += __shfl_xor(s,  off, 32);
        s2 += __shfl_xor(s2, off, 32);
    }
    __shared__ float ps[16], ps2[16];
    const int w = threadIdx.x >> 5, l = threadIdx.x & 31;
    if (l == 0) { ps[w] = s; ps2[w] = s2; }
    __syncthreads();
    if (w == 0) {
        float a  = (l < 16) ? ps[l]  : 0.0f;
        float a2 = (l < 16) ? ps2[l] : 0.0f;
        #pragma unroll
        for (int off = 8; off > 0; off >>= 1) {
            a  += __shfl_xor(a,  off, 32);
            a2 += __shfl_xor(a2, off, 32);
        }
        if (l == 0) { ps[0] = a; ps2[0] = a2; }
    }
    __syncthreads();

    const float mean = ps[0] * (1.0f / KDIM);
    const float var  = ps2[0] * (1.0f / KDIM) - mean * mean;
    const float xhat = (v - mean) * rsqrtf(var + 1e-5f);
    xn[idx] = (_Float16)(xhat * gamma[c] + beta[c]);
}

// ---------------------------------------------------------------------------
// Kernel 2: repack W [512,2048] fp32 -> f16 WMMA B-fragment layout
// Wp[kblk][nblk][lane][e], e=0..15 : K = kblk*32 + 8*(lane/16) + (e%8) + 16*(e/8),
//                                    N = nblk*16 + (lane%16)
// (16-bit B 32x16: same k order as A, upper lane-half holds K=8..15, 24..31)
// ---------------------------------------------------------------------------
__global__ __launch_bounds__(256) void pack_w(
    const float* __restrict__ W, _Float16* __restrict__ Wp)
{
    const int t    = blockIdx.x * 256 + threadIdx.x;   // 0 .. 65535
    const int kb   = t >> 12;                          // / (128*32)
    const int rem  = t & 4095;
    const int nb   = rem >> 5;
    const int lane = rem & 31;
    const int h    = lane >> 4;
    const int n    = lane & 15;

    v16h vv;
    #pragma unroll
    for (int e = 0; e < 16; ++e) {
        const int K = kb * 32 + 8 * h + (e & 7) + 16 * (e >> 3);
        const int N = nb * 16 + n;
        vv[e] = (_Float16)W[(size_t)K * NDIM + N];
    }
    *(v16h*)(Wp + (size_t)t * 16) = vv;
}

// ---------------------------------------------------------------------------
// Kernel 3: GEMM (WMMA f32_16x16x32_f16) + bias + exact GELU
// block = 256 threads = 8 waves; workgroup tile 64M x 256N
// wave grid 2x4, each wave: 32M x 64N (2 A-frags x 4 B-frags, 8 C-frags)
// ---------------------------------------------------------------------------
__global__ __launch_bounds__(256) void gemm_gelu(
    const _Float16* __restrict__ Xn, const _Float16* __restrict__ Wp,
    const float* __restrict__ bias, float* __restrict__ out)
{
    const int lane = threadIdx.x & 31;
    const int wave = threadIdx.x >> 5;     // 0..7
    const int wrow = wave >> 2;            // 0..1
    const int wcol = wave & 3;             // 0..3

    const int m0  = blockIdx.x * 64 + wrow * 32;     // row base of this wave
    const int n0  = blockIdx.y * 256 + wcol * 64;    // col base of this wave
    const int nb0 = n0 >> 4;

    const int half = lane >> 4;            // 0/1
    const int r    = lane & 15;

    v8f c[2][4] = {};

    // K loop: 512 / 32 = 16 steps, 8 WMMAs per step
    #pragma unroll 4
    for (int kb = 0; kb < KDIM / 32; ++kb) {
        // A fragments (16-bit A 16x32 layout):
        // lane-half 0: K = kb*32 + [0..7, 16..23]; lane-half 1: +8
        AFrag a[2];
        #pragma unroll
        for (int i = 0; i < 2; ++i) {
            const _Float16* p =
                Xn + (size_t)(m0 + i * 16 + r) * KDIM + kb * 32 + 8 * half;
            a[i].h[0] = *(const v8h*)(p);
            a[i].h[1] = *(const v8h*)(p + 16);
        }
        #pragma unroll
        for (int j = 0; j < 4; ++j) {
            const v16h bfrag = *(const v16h*)(
                Wp + ((size_t)(kb * (NDIM / 16) + nb0 + j) * 32 + lane) * 16);
            #pragma unroll
            for (int i = 0; i < 2; ++i) {
                c[i][j] = __builtin_amdgcn_wmma_f32_16x16x32_f16(
                    false, a[i].v, false, bfrag,
                    (short)0, c[i][j], false, false);
            }
        }
    }

    // Epilogue: bias + exact GELU.  D layout: M = q + 8*half, N = r
    #pragma unroll
    for (int j = 0; j < 4; ++j) {
        const int n  = n0 + j * 16 + r;
        const float bv = bias[n];
        #pragma unroll
        for (int i = 0; i < 2; ++i) {
            #pragma unroll
            for (int q = 0; q < 8; ++q) {
                const int m = m0 + i * 16 + q + 8 * half;
                float v = c[i][j][q] + bv;
                v = 0.5f * v * (1.0f + erff(v * 0.70710678118654752f));
                out[(size_t)m * NDIM + n] = v;
            }
        }
    }
}

// ---------------------------------------------------------------------------
// Launch
// inputs: 0=x203, 1=x217, 2=ln_gamma, 3=ln_beta, 4=W, 5=b   (all fp32)
// ws layout: [0, 12.85MB) f16 activations, then 2MB packed W
// ---------------------------------------------------------------------------
extern "C" void kernel_launch(void* const* d_in, const int* in_sizes, int n_in,
                              void* d_out, int out_size, void* d_ws, size_t ws_size,
                              hipStream_t stream) {
    const float* x203  = (const float*)d_in[0];
    const float* x217  = (const float*)d_in[1];
    const float* gamma = (const float*)d_in[2];
    const float* beta  = (const float*)d_in[3];
    const float* W     = (const float*)d_in[4];
    const float* bias  = (const float*)d_in[5];
    float* out = (float*)d_out;

    _Float16* xn = (_Float16*)d_ws;                                  // 12544*512*2 B
    _Float16* Wp = (_Float16*)((char*)d_ws + (size_t)MROWS * KDIM * 2); // 512*2048*2 B

    ln_kernel<<<MROWS, 512, 0, stream>>>(x203, x217, gamma, beta, xn);
    pack_w<<<(KDIM / 32) * (NDIM / 16) * 32 / 256 * 16 / 16, 256, 0, stream>>>(W, Wp); // 256 blocks
    gemm_gelu<<<dim3(MROWS / 64, NDIM / 256), 256, 0, stream>>>(xn, Wp, bias, out);
}
